// SlidingWindowAttention_9947144258211
// MI455X (gfx1250) — matrix operand order repaired and checked by reading, not compiled, not run on hardware
//
#include <hip/hip_runtime.h>

// ---------------------------------------------------------------------------
// Sliding-window attention for MI455X (gfx1250, wave32, WMMA).
// Pipeline: f32->f16 convert, QKV GEMM (wmma f16, f32 acc, f16 out),
// flash-style banded attention (wmma), output GEMM (wmma, f32 out).
// Row-major global->LDS staging uses GLOBAL_LOAD_ASYNC_TO_LDS_B128 (ASYNCcnt,
// CDNA5 ISA 15.18.3) with double-buffered pipelining in the GEMM.
// ---------------------------------------------------------------------------

typedef unsigned short u16;
typedef unsigned int   u32;
typedef __attribute__((ext_vector_type(16))) _Float16 v16h;
typedef __attribute__((ext_vector_type(8)))  float    v8f;

#define D_MODEL 1024
#define N_HEADS 16
#define D_HEAD  64
#define WIN     256
#define BATCH   4
#define SEQ     4096
#define NB      (SEQ / WIN)   // 16

__device__ __forceinline__ u16 f32_to_f16b(float f) {
  union { _Float16 h; u16 u; } v;
  v.h = (_Float16)f;
  return v.u;
}

__device__ __forceinline__ v8f zero8() {
  v8f r;
#pragma unroll
  for (int i = 0; i < 8; ++i) r[i] = 0.0f;
  return r;
}

// LDS byte offset of a __shared__ pointer (flat LDS addresses map via low 32b).
__device__ __forceinline__ u32 lds_off(const void* p) {
  return (u32)(size_t)p;
}

// Issue two async 16B global->LDS transfers (16 contiguous f16).
__device__ __forceinline__ void async_copy_32h(u32 lds, const u16* g) {
  asm volatile(
      "global_load_async_to_lds_b128 %0, %1, off\n\t"
      "global_load_async_to_lds_b128 %0, %1, off offset:16"
      :: "v"(lds), "v"(g) : "memory");
}

// Issue four async 16B global->LDS transfers (32 contiguous f16).
__device__ __forceinline__ void async_copy_64h(u32 lds, const u16* g) {
  asm volatile(
      "global_load_async_to_lds_b128 %0, %1, off\n\t"
      "global_load_async_to_lds_b128 %0, %1, off offset:16\n\t"
      "global_load_async_to_lds_b128 %0, %1, off offset:32\n\t"
      "global_load_async_to_lds_b128 %0, %1, off offset:48"
      :: "v"(lds), "v"(g) : "memory");
}

__device__ __forceinline__ void wait_async0() {
  asm volatile("s_wait_asynccnt 0x0" ::: "memory");
}

// A-fragment (16x32 f16, M x K): lane holds row m = lane&15.
// VGPR i (a dword = 2 f16): K = (i>=4 ? 16:0) + (lane>>4)*8 + (i&3)*2.
__device__ __forceinline__ v16h load_frag_a(const u16* base, int ld, int lane) {
  const int m = lane & 15, hi = lane >> 4;
  union { v16h v; u32 u[8]; } f;
  const u16* row = base + m * ld;
#pragma unroll
  for (int i = 0; i < 8; ++i) {
    const int k = ((i >= 4) ? 16 : 0) + hi * 8 + (i & 3) * 2;
    f.u[i] = *(const u32*)(row + k);
  }
  return f.v;
}

// B-fragment (32x16 f16, K x N) read from K-contiguous (transposed) storage:
// lane holds col n = lane&15; K = (i>=4 ? 16:0) + (lane>>4)*8 + (i&3)*2 + {0,1}.
__device__ __forceinline__ v16h load_frag_b(const u16* base, int ld, int lane) {
  const int n = lane & 15, hi = lane >> 4;
  union { v16h v; u32 u[8]; } f;
  const u16* row = base + n * ld;
#pragma unroll
  for (int i = 0; i < 8; ++i) f.u[i] = *(const u32*)(row + ((i >= 4) ? 16 : 0) + hi * 8 + (i & 3) * 2);
  return f.v;
}

__device__ __forceinline__ v8f wmma16(v16h a, v16h b, v8f c) {
  return __builtin_amdgcn_wmma_f32_16x16x32_f16(false, a, false, b,
                                                (short)0, c, false, false);
}

// ---------------------------------------------------------------------------
// Vectorized f32 -> f16-bits convert (n multiple of 4; true for all buffers).
__global__ void convert_f32_to_f16(const float* __restrict__ src,
                                   u16* __restrict__ dst, int n4) {
  int i = blockIdx.x * blockDim.x + threadIdx.x;
  const int stride = gridDim.x * blockDim.x;
  for (; i < n4; i += stride) {
    float4 f = ((const float4*)src)[i];
    u32 lo = (u32)f32_to_f16b(f.x) | ((u32)f32_to_f16b(f.y) << 16);
    u32 hi = (u32)f32_to_f16b(f.z) | ((u32)f32_to_f16b(f.w) << 16);
    ((u32*)dst)[2 * i]     = lo;
    ((u32*)dst)[2 * i + 1] = hi;
  }
}

// ---------------------------------------------------------------------------
// C[M,N] = A[M,K] @ B[K,N], A/B f16 row-major, f32 accumulate.
// Block tile 128x128, K-chunk 64, double-buffered async staging.
// 8 waves: 4 (M) x 2 (N), each computing 32x64.
template <int OUT_F16>
__global__ __launch_bounds__(256) void gemm_f16_wmma(
    const u16* __restrict__ A, const u16* __restrict__ Bm,
    void* __restrict__ Cout, const int M, const int N, const int K) {
  __shared__ alignas(16) u16 As[2][128 * 64];   // [m][k]
  __shared__ alignas(16) u16 Bs[2][128 * 64];   // [n][k] (transposed)

  const int tid = threadIdx.x;
  const int lane = tid & 31, wid = tid >> 5;
  const int wm = wid & 3, wn = wid >> 2;
  const int row0 = blockIdx.y * 128;
  const int col0 = blockIdx.x * 128;

  v8f acc[2][4];
#pragma unroll
  for (int tm = 0; tm < 2; ++tm)
#pragma unroll
    for (int tn = 0; tn < 4; ++tn) acc[tm][tn] = zero8();

  const int a_row = tid >> 1, a_seg = tid & 1;  // 128 rows x 2 segs of 32
  const int b_k   = tid >> 2, b_seg = tid & 3;  // 64 k-rows x 4 segs of 32
  const int a_off = a_row * 64 + a_seg * 32;
  const u16* a_g  = A + (size_t)(row0 + a_row) * K + a_seg * 32;
  const u16* b_g  = Bm + (size_t)b_k * N + col0 + b_seg * 32;
  const int nk = K / 64;

  // Prologue: stage chunk 0 into buffer 0.
  async_copy_64h(lds_off(&As[0][a_off]), a_g);
  {
    const uint4* src = (const uint4*)b_g;
    union { uint4 q[4]; u16 s[32]; } tb;
    tb.q[0] = src[0]; tb.q[1] = src[1]; tb.q[2] = src[2]; tb.q[3] = src[3];
#pragma unroll
    for (int i = 0; i < 32; ++i) Bs[0][(b_seg * 32 + i) * 64 + b_k] = tb.s[i];
  }
  wait_async0();
  __syncthreads();

  for (int kc = 0; kc < nk; ++kc) {
    const int buf = kc & 1;
    const bool has_next = (kc + 1 < nk);

    // Kick off next chunk: async A -> LDS[buf^1]; B -> VGPRs (stored after
    // compute so its global latency hides under the WMMAs).
    union { uint4 q[4]; u16 s[32]; } tb;
    if (has_next) {
      async_copy_64h(lds_off(&As[buf ^ 1][a_off]), a_g + (kc + 1) * 64);
      const uint4* src = (const uint4*)(b_g + (size_t)(kc + 1) * 64 * N);
      tb.q[0] = src[0]; tb.q[1] = src[1]; tb.q[2] = src[2]; tb.q[3] = src[3];
      if (kc + 2 < nk) {  // prefetch chunk k+2 -> global_prefetch_b8
        __builtin_prefetch(a_g + (kc + 2) * 64, 0, 0);
        __builtin_prefetch(b_g + (size_t)(kc + 2) * 64 * N, 0, 0);
      }
    }

    // Compute on buffer `buf`.
#pragma unroll
    for (int kk = 0; kk < 64; kk += 32) {
      v16h af[2], bfg[4];
#pragma unroll
      for (int tm = 0; tm < 2; ++tm)
        af[tm] = load_frag_a(&As[buf][(wm * 32 + tm * 16) * 64 + kk], 64, lane);
#pragma unroll
      for (int tn = 0; tn < 4; ++tn)
        bfg[tn] = load_frag_b(&Bs[buf][(wn * 64 + tn * 16) * 64 + kk], 64, lane);
#pragma unroll
      for (int tm = 0; tm < 2; ++tm)
#pragma unroll
        for (int tn = 0; tn < 4; ++tn)
          acc[tm][tn] = wmma16(af[tm], bfg[tn], acc[tm][tn]);
    }

    // Transpose-store next B tile, then one barrier per iteration:
    // stage wrote buf^1 (unread this iter), compute read buf.
    if (has_next) {
#pragma unroll
      for (int i = 0; i < 32; ++i)
        Bs[buf ^ 1][(b_seg * 32 + i) * 64 + b_k] = tb.s[i];
    }
    wait_async0();
    __syncthreads();
  }

  // C layout: lane holds col n=lane&15; VGPR r holds row r + 8*(lane>>4).
  const int cn = lane & 15, hi = lane >> 4;
#pragma unroll
  for (int tm = 0; tm < 2; ++tm)
#pragma unroll
    for (int tn = 0; tn < 4; ++tn)
#pragma unroll
      for (int r = 0; r < 8; ++r) {
        const int row = row0 + wm * 32 + tm * 16 + r + 8 * hi;
        const int col = col0 + wn * 64 + tn * 16 + cn;
        const float v = acc[tm][tn][r];
        if (OUT_F16)
          ((u16*)Cout)[(size_t)row * N + col] = f32_to_f16b(v);
        else
          ((float*)Cout)[(size_t)row * N + col] = v;
      }
}

// ---------------------------------------------------------------------------
// Banded attention. One WG (256 thr, 8 waves) per (block, head, batch).
// Each wave owns 32 query rows, processed as TWO FULLY SEQUENTIAL passes of
// 16 rows (outermost tm loop, not unrolled) so per-pass live state is
// O[4]+qf[2]+M/L ~ 64 VGPRs and the allocator stays under 256. The KV band
// is re-streamed per pass; the second pass hits L2 (192 MB >> qkv size).
// qkv layout (f16 bits): [(b*SEQ + l) * 3072 + s*1024 + h*64 + d].
__global__ __launch_bounds__(256) void swa_attention(
    const u16* __restrict__ qkv, u16* __restrict__ attn_out) {
  __shared__ alignas(16) u16 Ksm[64 * 64];       // [kv][d]   row-major
  __shared__ alignas(16) u16 Vtsm[64 * 64];      // [d][kv]   transposed
  __shared__ alignas(16) u16 Psm[8 * 16 * 64];   // per-wave P scratch [16][64]

  const int blk = blockIdx.x;   // 0..15
  const int h   = blockIdx.y;   // 0..15
  const int b   = blockIdx.z;   // 0..3
  const int tid = threadIdx.x;
  const int lane = tid & 31, w = tid >> 5;
  const int cn = lane & 15, hi = lane >> 4;

  const float scale = 0.125f;  // 1/sqrt(64)
  const int c0 = (blk == 0) ? 4 : 0;  // block 0: zero-padded half fully masked
  const int kv_row = tid >> 2, kv_seg = tid & 3;  // 64 rows x 4 segs of 16
  const u32 k_lds = lds_off(Ksm + kv_row * 64 + kv_seg * 16);
  u16* Pw = Psm + w * (16 * 64);

#pragma unroll 1
  for (int tm = 0; tm < 2; ++tm) {
    // Q fragments for this pass's 16 rows (2 K-steps).
    v16h qf[2];
#pragma unroll
    for (int ks = 0; ks < 2; ++ks) {
      const u16* base = qkv + (size_t)(b * SEQ + blk * WIN + w * 32 + tm * 16) * 3072
                        + h * D_HEAD + ks * 32;
      qf[ks] = load_frag_a(base, 3072, lane);
    }

    v8f O[4];
    float Mx[8], Ls[8];
#pragma unroll
    for (int tn = 0; tn < 4; ++tn) O[tn] = zero8();
#pragma unroll
    for (int r = 0; r < 8; ++r) { Mx[r] = -1.0e30f; Ls[r] = 0.0f; }

    for (int c = c0; c < 8; ++c) {
      // Stage K (async, row-major) and V (transposed via VGPRs) into LDS.
      {
        const int j = c * 64 + kv_row;  // position in 512-wide band
        const int l = (j < WIN) ? (blk - 1) * WIN + j : blk * WIN + (j - WIN);
        const size_t rowbase = (size_t)(b * SEQ + l) * 3072 + h * D_HEAD + kv_seg * 16;
        async_copy_32h(k_lds, qkv + rowbase + 1024);              // K chunk
        {
          const uint4* src = (const uint4*)(qkv + rowbase + 2048);  // V chunk
          union { uint4 q[2]; u16 s[16]; } tv;
          tv.q[0] = src[0]; tv.q[1] = src[1];
#pragma unroll
          for (int i = 0; i < 16; ++i)
            Vtsm[(kv_seg * 16 + i) * 64 + kv_row] = tv.s[i];
        }
      }
      wait_async0();
      __syncthreads();

      // S = Q @ K^T for this 16x64 strip.
      v8f S[4];
#pragma unroll
      for (int tn = 0; tn < 4; ++tn) S[tn] = zero8();
#pragma unroll
      for (int ks = 0; ks < 2; ++ks) {
        v16h kf[4];
#pragma unroll
        for (int tn = 0; tn < 4; ++tn)
          kf[tn] = load_frag_b(Ksm + (tn * 16) * 64 + ks * 32, 64, lane);
#pragma unroll
        for (int tn = 0; tn < 4; ++tn)
          S[tn] = wmma16(qf[ks], kf[tn], S[tn]);
      }

      // Scale + band mask + online softmax; write P (f16) to wave scratch.
#pragma unroll
      for (int r = 0; r < 8; ++r) {
        const int ti = w * 32 + tm * 16 + r + 8 * hi;  // query pos in block
        float sv[4];
        float smax = -1.0e30f;
#pragma unroll
        for (int tn = 0; tn < 4; ++tn) {
          const int j = c * 64 + tn * 16 + cn;
          float s = S[tn][r] * scale;
          const bool valid = (j >= ti + 1) && (j <= ti + WIN);
          s = valid ? s : -1.0e30f;
          sv[tn] = s;
          smax = fmaxf(smax, s);
        }
        // Butterfly over 16 lanes (lower/upper halves reduce independently,
        // matching the C-layout row split r / r+8).
#pragma unroll
        for (int off = 1; off < 16; off <<= 1)
          smax = fmaxf(smax, __shfl_xor(smax, off, 32));
        const float mnew  = fmaxf(Mx[r], smax);
        const float alpha = __expf(Mx[r] - mnew);
        Mx[r] = mnew;
        float psum = 0.0f;
#pragma unroll
        for (int tn = 0; tn < 4; ++tn) {
          const float p = __expf(sv[tn] - mnew);
          psum += p;
          Pw[(r + 8 * hi) * 64 + tn * 16 + cn] = f32_to_f16b(p);
        }
#pragma unroll
        for (int off = 1; off < 16; off <<= 1)
          psum += __shfl_xor(psum, off, 32);
        Ls[r] = Ls[r] * alpha + psum;
#pragma unroll
        for (int tn = 0; tn < 4; ++tn) O[tn][r] *= alpha;
      }

      // O += P @ V  (P re-laid-out via per-wave LDS round trip; LDS ops of a
      // wave are in order, so reuse of Pw across chunks/passes is safe).
#pragma unroll
      for (int ks = 0; ks < 2; ++ks) {
        v16h pf = load_frag_a(Pw + ks * 32, 64, lane);
        v16h vf[4];
#pragma unroll
        for (int tn = 0; tn < 4; ++tn)
          vf[tn] = load_frag_b(Vtsm + (tn * 16) * 64 + ks * 32, 64, lane);
#pragma unroll
        for (int tn = 0; tn < 4; ++tn)
          O[tn] = wmma16(pf, vf[tn], O[tn]);
      }
      __syncthreads();
    }

    // Normalize and emit f16 rows for this pass.
#pragma unroll
    for (int r = 0; r < 8; ++r) {
      const float inv = 1.0f / Ls[r];
      const int ti = w * 32 + tm * 16 + r + 8 * hi;
      const size_t rowb = (size_t)(b * SEQ + blk * WIN + ti) * D_MODEL + h * D_HEAD;
#pragma unroll
      for (int tn = 0; tn < 4; ++tn)
        attn_out[rowb + tn * 16 + cn] = f32_to_f16b(O[tn][r] * inv);
    }
  }
}

// ---------------------------------------------------------------------------
extern "C" void kernel_launch(void* const* d_in, const int* in_sizes, int n_in,
                              void* d_out, int out_size, void* d_ws, size_t ws_size,
                              hipStream_t stream) {
  const float* x     = (const float*)d_in[0];
  const float* w_qkv = (const float*)d_in[1];
  const float* w_out = (const float*)d_in[2];
  float* out = (float*)d_out;

  // Workspace partition (all f16-bit buffers), ~168 MB total.
  char* ws = (char*)d_ws;
  u16* x_f16    = (u16*)ws; ws += (size_t)BATCH * SEQ * D_MODEL * 2;       // 32 MB
  u16* wqkv_f16 = (u16*)ws; ws += (size_t)D_MODEL * 3 * D_MODEL * 2;       // 6 MB
  u16* wout_f16 = (u16*)ws; ws += (size_t)D_MODEL * D_MODEL * 2;           // 2 MB
  u16* qkv_f16  = (u16*)ws; ws += (size_t)BATCH * SEQ * 3 * D_MODEL * 2;   // 96 MB
  u16* attn_f16 = (u16*)ws;                                                // 32 MB

  convert_f32_to_f16<<<2048, 256, 0, stream>>>(x, x_f16, BATCH * SEQ * D_MODEL / 4);
  convert_f32_to_f16<<<512, 256, 0, stream>>>(w_qkv, wqkv_f16, D_MODEL * 3 * D_MODEL / 4);
  convert_f32_to_f16<<<256, 256, 0, stream>>>(w_out, wout_f16, D_MODEL * D_MODEL / 4);

  // QKV projection: (16384 x 1024) @ (1024 x 3072) -> f16
  gemm_f16_wmma<1><<<dim3(3 * D_MODEL / 128, BATCH * SEQ / 128), 256, 0, stream>>>(
      x_f16, wqkv_f16, (void*)qkv_f16, BATCH * SEQ, 3 * D_MODEL, D_MODEL);

  // Banded attention, one WG per (block, head, batch).
  swa_attention<<<dim3(NB, N_HEADS, BATCH), 256, 0, stream>>>(qkv_f16, attn_f16);

  // Output projection: (16384 x 1024) @ (1024 x 1024) -> f32 d_out
  gemm_f16_wmma<0><<<dim3(D_MODEL / 128, BATCH * SEQ / 128), 256, 0, stream>>>(
      attn_f16, wout_f16, (void*)out, BATCH * SEQ, D_MODEL, D_MODEL);
}
